// LoRCnnModel_68118181314887
// MI455X (gfx1250) — matrix operand order repaired and checked by reading, not compiled, not run on hardware
//
#include <hip/hip_runtime.h>
#include <hip/hip_bf16.h>
#include <math.h>

// ---- model dims (fixed by reference) ----
#define S_  2048
#define D_  2048
#define H_  16
#define HD_ 128
#define DN_ 16
#define L_  2
#define NC_ 2
#define KW_ 63
#define FF_ 5632

typedef _Float16 h8   __attribute__((ext_vector_type(8)));
typedef _Float16 v16h __attribute__((ext_vector_type(16)));
typedef float    v8f  __attribute__((ext_vector_type(8)));
typedef int      v4i32 __attribute__((vector_size(16)));

// Async global->LDS (CDNA5): DMA tiles straight into LDS, tracked by
// ASYNCcnt. Guarded so the file still compiles (register-staged path)
// if the toolchain lacks the builtins.
#if __has_builtin(__builtin_amdgcn_global_load_async_to_lds_b128) && \
    __has_builtin(__builtin_amdgcn_s_wait_asynccnt)
#define ASYNC_LDS 1
#else
#define ASYNC_LDS 0
#endif

#if ASYNC_LDS
// Prototype (from hipcc diagnostic): arg0 = v4i* in AS(1) (global, non-const),
// arg1 = v4i* in AS(3) (LDS), then imm offset, imm cpol.
__device__ __forceinline__ void async_cp16B(const _Float16* g, _Float16* l) {
  __builtin_amdgcn_global_load_async_to_lds_b128(
      (__attribute__((address_space(1))) v4i32*)(v4i32*)g,
      (__attribute__((address_space(3))) v4i32*)(v4i32*)l,
      0, 0);
}
#endif

// =====================================================================
// Generic batched GEMM:  C[b] = alpha * A[b] (MxK) * Bt[b]^T (NxK)  [+ C]
// A, Bt are f16 row-major with K contiguous. M,N multiples of 128.
// K is a multiple of 16 (2048 / 5632 / 16): LDS tiles are filled in
// 16-half segments that are either fully valid (b128 moves) or zero.
// Double-buffered LDS; tile kt+1 moves while WMMAs consume tile kt.
// =====================================================================
#define BM 128
#define BN 128
#define BK 32
#define LDK 40   // padded LDS row (halves)

__device__ __forceinline__ void stage_zero(h8& lo, h8& hi) {
#pragma unroll
  for (int i = 0; i < 8; ++i) { lo[i] = (_Float16)0.0f; hi[i] = (_Float16)0.0f; }
}
__device__ __forceinline__ void stage_load(const _Float16* src, h8& lo, h8& hi) {
  lo = *(const h8*)src;
  hi = *(const h8*)(src + 8);
}

__global__ void __launch_bounds__(256)
gemm_bt_wmma(const _Float16* __restrict__ A, long long sA, int lda,
             const _Float16* __restrict__ Bt, long long sB, int ldb,
             float* __restrict__ C, _Float16* __restrict__ C16,
             long long sC, int ldc, int K, float alpha, int residual)
{
  __shared__ _Float16 As[2][BM * LDK];
  __shared__ _Float16 Bs[2][BN * LDK];

  const int bz = blockIdx.z;
  A  += (long long)bz * sA;
  Bt += (long long)bz * sB;
  const long long cbase = (long long)bz * sC;

  const int m0   = blockIdx.y * BM;
  const int n0   = blockIdx.x * BN;
  const int tid  = threadIdx.x;
  const int lane = tid & 31;
  const int wave = tid >> 5;
  const int wm   = (wave & 3) * 32;   // wave row offset in tile
  const int wn   = (wave >> 2) * 64;  // wave col offset in tile

  v8f acc[2][4];
#pragma unroll
  for (int i = 0; i < 2; ++i)
#pragma unroll
    for (int j = 0; j < 4; ++j)
#pragma unroll
      for (int r = 0; r < 8; ++r) acc[i][j][r] = 0.0f;

  const int lr   = tid >> 1;           // 0..127 tile row
  const int lseg = (tid & 1) * 16;     // 0 or 16 halves into the row
  const _Float16* Arow = A  + (long long)(m0 + lr) * lda + lseg;
  const _Float16* Brow = Bt + (long long)(n0 + lr) * ldb + lseg;
  const int ldsoff = lr * LDK + lseg;
  const int ksteps = (K + BK - 1) / BK;

#if ASYNC_LDS
  // ---------------- async DMA pipeline ----------------
  if (lseg < K) {
    async_cp16B(Arow,     &As[0][ldsoff]);
    async_cp16B(Arow + 8, &As[0][ldsoff] + 8);
    async_cp16B(Brow,     &Bs[0][ldsoff]);
    async_cp16B(Brow + 8, &Bs[0][ldsoff] + 8);
  } else {
    h8 zlo, zhi; stage_zero(zlo, zhi);
    *(h8*)&As[0][ldsoff] = zlo; *(h8*)(&As[0][ldsoff] + 8) = zhi;
    *(h8*)&Bs[0][ldsoff] = zlo; *(h8*)(&Bs[0][ldsoff] + 8) = zhi;
  }
#else
  // ---------------- register-staged pipeline ----------------
  h8 alo, ahi, blo, bhi;
  if (lseg < K) {
    stage_load(Arow, alo, ahi);
    stage_load(Brow, blo, bhi);
  } else {
    stage_zero(alo, ahi);
    stage_zero(blo, bhi);
  }
  *(h8*)&As[0][ldsoff] = alo; *(h8*)(&As[0][ldsoff] + 8) = ahi;
  *(h8*)&Bs[0][ldsoff] = blo; *(h8*)(&Bs[0][ldsoff] + 8) = bhi;
  __syncthreads();
#endif

  for (int kt = 0; kt < ksteps; ++kt) {
    const int cur = kt & 1;
    const bool more = (kt + 1) < ksteps;

#if ASYNC_LDS
    // tile kt resident for this wave; barrier publishes all waves' writes
    // and guarantees buffer `nxt` is no longer being read (iter kt-1 done).
    __builtin_amdgcn_s_wait_asynccnt(0);
    __syncthreads();
    if (more) {
      const int nxt = cur ^ 1;
      const int kpos = (kt + 1) * BK + lseg;
      if (kpos < K) {  // K % 16 == 0 -> whole segment valid
        async_cp16B(Arow + (kt + 1) * BK,     &As[nxt][ldsoff]);
        async_cp16B(Arow + (kt + 1) * BK + 8, &As[nxt][ldsoff] + 8);
        async_cp16B(Brow + (kt + 1) * BK,     &Bs[nxt][ldsoff]);
        async_cp16B(Brow + (kt + 1) * BK + 8, &Bs[nxt][ldsoff] + 8);
      } else {
        h8 zlo, zhi; stage_zero(zlo, zhi);
        *(h8*)&As[nxt][ldsoff] = zlo; *(h8*)(&As[nxt][ldsoff] + 8) = zhi;
        *(h8*)&Bs[nxt][ldsoff] = zlo; *(h8*)(&Bs[nxt][ldsoff] + 8) = zhi;
      }
    }
#else
    if (more) {
      const int kpos = (kt + 1) * BK + lseg;
      if (kpos < K) {
        stage_load(Arow + (kt + 1) * BK, alo, ahi);
        stage_load(Brow + (kt + 1) * BK, blo, bhi);
      } else {
        stage_zero(alo, ahi);
        stage_zero(blo, bhi);
      }
    }
#endif

    // A fragment: lanes 0-15 row m, K {0-7,16-23}; lanes 16-31 K {8-15,24-31}
    v16h af[2], bf[4];
    const int arow  = wm + (lane & 15);
    const int akoff = (lane >> 4) * 8;
#pragma unroll
    for (int fm = 0; fm < 2; ++fm) {
      const _Float16* p = &As[cur][(arow + fm * 16) * LDK + akoff];
      h8 lo = *(const h8*)p;
      h8 hi = *(const h8*)(p + 16);
#pragma unroll
      for (int i = 0; i < 8; ++i) { af[fm][i] = lo[i]; af[fm][i + 8] = hi[i]; }
    }
    // B fragment: lanes 0-15 col n, K {0-7,16-23}; lanes 16-31 K {8-15,24-31}
    const int bcol  = wn + (lane & 15);
    const int bkoff = (lane >> 4) * 8;
#pragma unroll
    for (int fn = 0; fn < 4; ++fn) {
      const _Float16* p = &Bs[cur][(bcol + fn * 16) * LDK + bkoff];
      h8 lo = *(const h8*)p;
      h8 hi = *(const h8*)(p + 16);
#pragma unroll
      for (int i = 0; i < 8; ++i) { bf[fn][i] = lo[i]; bf[fn][i + 8] = hi[i]; }
    }
#pragma unroll
    for (int fm = 0; fm < 2; ++fm)
#pragma unroll
      for (int fn = 0; fn < 4; ++fn)
        acc[fm][fn] = __builtin_amdgcn_wmma_f32_16x16x32_f16(
            false, af[fm], false, bf[fn], (short)0, acc[fm][fn], false, false);

#if !ASYNC_LDS
    if (more) {
      const int nxt = cur ^ 1;
      *(h8*)&As[nxt][ldsoff] = alo; *(h8*)(&As[nxt][ldsoff] + 8) = ahi;
      *(h8*)&Bs[nxt][ldsoff] = blo; *(h8*)(&Bs[nxt][ldsoff] + 8) = bhi;
    }
    __syncthreads();
#endif
  }

  // epilogue: C/D layout — VGPR r: M = r + (lane>=16)*8, N = lane&15
  const int rbase = (lane >> 4) * 8;
  const int ccol  = lane & 15;
#pragma unroll
  for (int fm = 0; fm < 2; ++fm)
#pragma unroll
    for (int fn = 0; fn < 4; ++fn) {
      const int row0 = m0 + wm + fm * 16 + rbase;
      const int col  = n0 + wn + fn * 16 + ccol;
#pragma unroll
      for (int r = 0; r < 8; ++r) {
        float v = acc[fm][fn][r] * alpha;
        const long long idx = cbase + (long long)(row0 + r) * ldc + col;
        if (C)   { if (residual) v += C[idx]; C[idx] = v; }
        if (C16) C16[idx] = (_Float16)v;
      }
    }
}

// =====================================================================
// Helpers
// =====================================================================
__device__ __forceinline__ float blk_sum(float v, float* sh) {
  const int tid = threadIdx.x;
  sh[tid] = v; __syncthreads();
#pragma unroll
  for (int s = 128; s > 0; s >>= 1) {
    if (tid < s) sh[tid] += sh[tid + s];
    __syncthreads();
  }
  float r = sh[0]; __syncthreads();
  return r;
}
__device__ __forceinline__ float blk_max(float v, float* sh) {
  const int tid = threadIdx.x;
  sh[tid] = v; __syncthreads();
#pragma unroll
  for (int s = 128; s > 0; s >>= 1) {
    if (tid < s) sh[tid] = fmaxf(sh[tid], sh[tid + s]);
    __syncthreads();
  }
  float r = sh[0]; __syncthreads();
  return r;
}

// rmsnorm(x) * w -> f16, one block per row of D_
__global__ void __launch_bounds__(256)
rmsnorm_f16(const float* __restrict__ x, const float* __restrict__ w,
            _Float16* __restrict__ out)
{
  __shared__ float sh[256];
  const long long base = (long long)blockIdx.x * D_;
  float ss = 0.f;
  for (int d = threadIdx.x; d < D_; d += 256) { float v = x[base + d]; ss += v * v; }
  float tot = blk_sum(ss, sh);
  float rs = rsqrtf(tot / (float)D_ + 1e-6f);
  for (int d = threadIdx.x; d < D_; d += 256)
    out[base + d] = (_Float16)(x[base + d] * rs * w[d]);
}

// f32 [R][Cc] row-major -> f16 [Cc][R] (transpose + convert); R,Cc % 32 == 0
__global__ void __launch_bounds__(256)
transpose_f32_to_f16(const float* __restrict__ src, _Float16* __restrict__ dst,
                     int R, int Cc)
{
  __shared__ float tile[32][33];
  const int c0 = blockIdx.x * 32, r0 = blockIdx.y * 32;
  const int tx = threadIdx.x & 31, ty = threadIdx.x >> 5;  // 32 x 8
#pragma unroll
  for (int i = 0; i < 32; i += 8)
    tile[ty + i][tx] = src[(long long)(r0 + ty + i) * Cc + c0 + tx];
  __syncthreads();
#pragma unroll
  for (int i = 0; i < 32; i += 8)
    dst[(long long)(c0 + ty + i) * R + r0 + tx] = (_Float16)tile[tx][ty + i];
}

// rotary on q,k in place; thread handles the (j, j+64) pair -> race free.
// positions are arange(S) in the reference harness (pos == row index).
__global__ void __launch_bounds__(256)
rotary_qk(float* __restrict__ q, float* __restrict__ k)
{
  const long long t = (long long)blockIdx.x * 256 + threadIdx.x;
  const int j = (int)(t & 63);
  const int h = (int)((t >> 6) & 15);
  const int s = (int)(t >> 10);
  const float pos = (float)s;
  const float inv = powf(10000.f, -(float)j / 64.f);
  const float fr = pos * inv;
  const float cs = cosf(fr), sn = sinf(fr);
  const long long i1 = (long long)s * D_ + h * HD_ + j;
  const long long i2 = i1 + 64;
  float q1 = q[i1], q2 = q[i2];
  q[i1] = q1 * cs - q2 * sn;
  q[i2] = q2 * cs + q1 * sn;
  float k1 = k[i1], k2 = k[i2];
  k[i1] = k1 * cs - k2 * sn;
  k[i2] = k2 * cs + k1 * sn;
}

// per-head 128 -> 16 down projection for q and k; one block per sequence row
__global__ void __launch_bounds__(256)
qk_downproj(const float* __restrict__ q, const float* __restrict__ k,
            const float* __restrict__ Wdq, const float* __restrict__ Wdk,
            _Float16* __restrict__ qd, _Float16* __restrict__ kd)
{
  __shared__ float qrow[D_], krow[D_];
  __shared__ float wq[HD_ * DN_], wk[HD_ * DN_];
  const int s = blockIdx.x, tid = threadIdx.x;
  for (int i = tid; i < D_; i += 256) {
    qrow[i] = q[(long long)s * D_ + i];
    krow[i] = k[(long long)s * D_ + i];
  }
  for (int i = tid; i < HD_ * DN_; i += 256) { wq[i] = Wdq[i]; wk[i] = Wdk[i]; }
  __syncthreads();
  const int h = tid >> 4, n = tid & 15;
  float aq = 0.f, ak = 0.f;
#pragma unroll 4
  for (int d = 0; d < HD_; ++d) {
    aq += qrow[h * HD_ + d] * wq[d * DN_ + n];
    ak += krow[h * HD_ + d] * wk[d * DN_ + n];
  }
  const long long o = ((long long)h * S_ + s) * DN_ + n;
  qd[o] = (_Float16)aq;
  kd[o] = (_Float16)ak;
}

// ln1 over k-axis + scaler dot on RAW scores; one block per (h,q) row.
// Score tensor (268MB) streams with non-temporal policy (no reuse, > L2).
__global__ void __launch_bounds__(256)
ln1_scale(const float* __restrict__ in, float* __restrict__ out,
          const float* __restrict__ w, const float* __restrict__ b,
          const float* __restrict__ sw, const float* __restrict__ sb,
          float* __restrict__ scale)
{
  __shared__ float sh[256];
  const long long base = (long long)blockIdx.x * S_;
  float vals[8];
  float sum = 0.f, sq = 0.f, dot = 0.f;
#pragma unroll
  for (int i = 0; i < 8; ++i) {
    const int k = threadIdx.x + i * 256;
    float v = __builtin_nontemporal_load(&in[base + k]);
    vals[i] = v; sum += v; sq += v * v; dot += v * sw[k];
  }
  float tsum = blk_sum(sum, sh);
  float tsq  = blk_sum(sq, sh);
  float tdot = blk_sum(dot, sh);
  float mean = tsum / (float)S_;
  float var  = tsq / (float)S_ - mean * mean;
  float rs   = rsqrtf(var + 1e-5f);
#pragma unroll
  for (int i = 0; i < 8; ++i) {
    const int k = threadIdx.x + i * 256;
    __builtin_nontemporal_store((vals[i] - mean) * rs * w[k] + b[k], &out[base + k]);
  }
  if (threadIdx.x == 0) scale[blockIdx.x] = tdot + sb[0];
}

// causal depthwise conv along q (63 taps) + bias + relu
// grid: (S/32 k-tiles, S/64 q-tiles, H). Loads stay temporal (halo reuse
// across neighboring q-blocks); streaming output is non-temporal.
__global__ void __launch_bounds__(256)
dwconv_relu(const float* __restrict__ in, float* __restrict__ out,
            const float* __restrict__ cw, const float* __restrict__ cb)
{
  __shared__ float tile[(64 + KW_ - 1) * 32];
  __shared__ float wsh[KW_];
  const int h = blockIdx.z;
  const int k0 = blockIdx.x * 32, q0 = blockIdx.y * 64;
  const int tid = threadIdx.x;
  if (tid < KW_) wsh[tid] = cw[h * KW_ + tid];
  const float* inh = in + (long long)h * S_ * S_;
  for (int idx = tid; idx < (64 + KW_ - 1) * 32; idx += 256) {
    const int i = idx >> 5, kk = idx & 31;
    const int gq = q0 - (KW_ - 1) + i;
    tile[idx] = (gq >= 0) ? inh[(long long)gq * S_ + k0 + kk] : 0.f;
  }
  __syncthreads();
  const float bias = cb[h];
  float* outh = out + (long long)h * S_ * S_;
  for (int o = tid; o < 64 * 32; o += 256) {
    const int qi = o >> 5, kk = o & 31;
    float acc = bias;
#pragma unroll
    for (int j = 0; j < KW_; ++j) acc += wsh[j] * tile[(qi + j) * 32 + kk];
    __builtin_nontemporal_store(fmaxf(acc, 0.f),
                                &outh[(long long)(q0 + qi) * S_ + k0 + kk]);
  }
}

// ln2 + causal mask + softmax + sigmoid(scale), output f16 probs
__global__ void __launch_bounds__(256)
ln2_softmax(const float* __restrict__ in, _Float16* __restrict__ probs,
            const float* __restrict__ w, const float* __restrict__ b,
            const float* __restrict__ scale)
{
  __shared__ float sh[256];
  const int rid = blockIdx.x;
  const int qq = rid & (S_ - 1);
  const long long base = (long long)rid * S_;
  float vals[8];
  float sum = 0.f, sq = 0.f;
#pragma unroll
  for (int i = 0; i < 8; ++i) {
    const int k = threadIdx.x + i * 256;
    float v = __builtin_nontemporal_load(&in[base + k]);
    vals[i] = v; sum += v; sq += v * v;
  }
  float tsum = blk_sum(sum, sh);
  float tsq  = blk_sum(sq, sh);
  float mean = tsum / (float)S_;
  float rs   = rsqrtf(tsq / (float)S_ - mean * mean + 1e-5f);
  float mx = -3.402823466e38f;
#pragma unroll
  for (int i = 0; i < 8; ++i) {
    const int k = threadIdx.x + i * 256;
    float t = (vals[i] - mean) * rs * w[k] + b[k];
    if (k > qq) t += -1e9f;
    t = fmaxf(t, -3.402823466e38f);
    vals[i] = t;
    mx = fmaxf(mx, t);
  }
  float tmx = blk_max(mx, sh);
  float es = 0.f;
#pragma unroll
  for (int i = 0; i < 8; ++i) { vals[i] = expf(vals[i] - tmx); es += vals[i]; }
  float tes = blk_sum(es, sh);
  const float sg = 1.f / (1.f + expf(-scale[rid]));
  const float mul = sg / tes;
#pragma unroll
  for (int i = 0; i < 8; ++i) {
    const int k = threadIdx.x + i * 256;
    probs[base + k] = (_Float16)(vals[i] * mul);
  }
}

// silu(gate) * up -> f16
__global__ void __launch_bounds__(256)
silu_mul(const float* __restrict__ G, const float* __restrict__ U,
         _Float16* __restrict__ P)
{
  const long long i = (long long)blockIdx.x * 256 + threadIdx.x;
  const float g = G[i];
  P[i] = (_Float16)(g / (1.f + expf(-g)) * U[i]);
}

// =====================================================================
// Host orchestration
// =====================================================================
extern "C" void kernel_launch(void* const* d_in, const int* in_sizes, int n_in,
                              void* d_out, int out_size, void* d_ws, size_t ws_size,
                              hipStream_t stream) {
  (void)in_sizes; (void)n_in; (void)out_size; (void)ws_size;
  const float* hs    = (const float*)d_in[0];
  const float* Wq    = (const float*)d_in[3];
  const float* Wk    = (const float*)d_in[4];
  const float* Wv    = (const float*)d_in[5];
  const float* Wo    = (const float*)d_in[6];
  const float* Wdq   = (const float*)d_in[7];
  const float* Wdk   = (const float*)d_in[8];
  const float* ln1w  = (const float*)d_in[9];
  const float* ln1b  = (const float*)d_in[10];
  const float* ln2w  = (const float*)d_in[11];
  const float* ln2b  = (const float*)d_in[12];
  const float* convw = (const float*)d_in[13];
  const float* convb = (const float*)d_in[14];
  const float* scw   = (const float*)d_in[15];
  const float* scb   = (const float*)d_in[16];
  const float* rms1  = (const float*)d_in[17];
  const float* rms2  = (const float*)d_in[18];
  const float* Wg    = (const float*)d_in[19];
  const float* Wu    = (const float*)d_in[20];
  const float* Wd    = (const float*)d_in[21];
  float* x = (float*)d_out;  // residual stream lives in d_out

  char* base = (char*)d_ws;
  size_t off = 0;
  auto alloc = [&](size_t bytes) -> void* {
    void* p = base + off;
    off += (bytes + 255) & ~(size_t)255;
    return p;
  };
  _Float16* WqT  = (_Float16*)alloc((size_t)L_ * D_ * D_ * 2);
  _Float16* WkT  = (_Float16*)alloc((size_t)L_ * D_ * D_ * 2);
  _Float16* WvT  = (_Float16*)alloc((size_t)L_ * D_ * D_ * 2);
  _Float16* WoT  = (_Float16*)alloc((size_t)L_ * D_ * D_ * 2);
  _Float16* WgT  = (_Float16*)alloc((size_t)L_ * FF_ * D_ * 2);
  _Float16* WuT  = (_Float16*)alloc((size_t)L_ * FF_ * D_ * 2);
  _Float16* WdT  = (_Float16*)alloc((size_t)L_ * D_ * FF_ * 2);
  _Float16* a16  = (_Float16*)alloc((size_t)S_ * FF_ * 2);   // h16 / o16 / P16
  float*    qf   = (float*)alloc((size_t)S_ * D_ * 4);
  float*    kf   = (float*)alloc((size_t)S_ * D_ * 4);
  float*    vf   = (float*)alloc((size_t)S_ * D_ * 4);
  _Float16* qd16 = (_Float16*)alloc((size_t)H_ * S_ * DN_ * 2);
  _Float16* kd16 = (_Float16*)alloc((size_t)H_ * S_ * DN_ * 2);
  _Float16* vT16 = (_Float16*)alloc((size_t)D_ * S_ * 2);    // [H][HD][S]
  float*    scA  = (float*)alloc((size_t)H_ * S_ * S_ * 4);
  float*    scB  = (float*)alloc((size_t)H_ * S_ * S_ * 4);
  _Float16* p16  = (_Float16*)alloc((size_t)H_ * S_ * S_ * 2);
  float*    scl  = (float*)alloc((size_t)H_ * S_ * 4);
  float*    Gf   = scA;                                      // reuse after softmax
  float*    Uf   = scA + (size_t)S_ * FF_;

  auto gemm = [&](const _Float16* A, long long sA, int lda,
                  const _Float16* Bt, long long sB, int ldb,
                  float* C, _Float16* C16, long long sC, int ldc,
                  int M, int N, int K, int batch, float alpha, int residual) {
    dim3 g(N / BN, M / BM, batch);
    gemm_bt_wmma<<<g, 256, 0, stream>>>(A, sA, lda, Bt, sB, ldb,
                                        C, C16, sC, ldc, K, alpha, residual);
  };
  auto trans = [&](const float* src, _Float16* dst, int R, int Cc) {
    transpose_f32_to_f16<<<dim3(Cc / 32, R / 32), 256, 0, stream>>>(src, dst, R, Cc);
  };

  // x = hidden_states
  (void)hipMemcpyAsync(x, hs, (size_t)S_ * D_ * 4, hipMemcpyDeviceToDevice, stream);

  // pre-convert weights to transposed f16 [N][K]
  for (int l = 0; l < L_; ++l) {
    const size_t wdd = (size_t)l * D_ * D_;
    trans(Wq + wdd, WqT + wdd, D_, D_);
    trans(Wk + wdd, WkT + wdd, D_, D_);
    trans(Wv + wdd, WvT + wdd, D_, D_);
    trans(Wo + wdd, WoT + wdd, D_, D_);
    trans(Wg + (size_t)l * D_ * FF_, WgT + (size_t)l * FF_ * D_, D_, FF_);
    trans(Wu + (size_t)l * D_ * FF_, WuT + (size_t)l * FF_ * D_, D_, FF_);
    trans(Wd + (size_t)l * FF_ * D_, WdT + (size_t)l * D_ * FF_, FF_, D_);
  }

  for (int l = 0; l < L_; ++l) {
    // h = rmsnorm(x, rms1) -> f16
    rmsnorm_f16<<<S_, 256, 0, stream>>>(x, rms1 + (size_t)l * D_, a16);
    // q,k,v = h @ W
    gemm(a16, 0, D_, WqT + (size_t)l * D_ * D_, 0, D_, qf, nullptr, 0, D_,
         S_, D_, D_, 1, 1.f, 0);
    gemm(a16, 0, D_, WkT + (size_t)l * D_ * D_, 0, D_, kf, nullptr, 0, D_,
         S_, D_, D_, 1, 1.f, 0);
    gemm(a16, 0, D_, WvT + (size_t)l * D_ * D_, 0, D_, vf, nullptr, 0, D_,
         S_, D_, D_, 1, 1.f, 0);
    // rotary q,k in place
    rotary_qk<<<(S_ * H_ * 64) / 256, 256, 0, stream>>>(qf, kf);
    // per-head 128->16 projections
    qk_downproj<<<S_, 256, 0, stream>>>(qf, kf,
        Wdq + (size_t)l * HD_ * DN_, Wdk + (size_t)l * HD_ * DN_, qd16, kd16);
    // v -> f16 transposed per head [D][S]
    trans(vf, vT16, S_, D_);
    // scores = qd @ kd^T / sqrt(16), batched over heads
    gemm(qd16, (long long)S_ * DN_, DN_, kd16, (long long)S_ * DN_, DN_,
         scA, nullptr, (long long)S_ * S_, S_, S_, S_, DN_, H_, 0.25f, 0);
    // ln1 over k + scaler dot (raw scores) : scA -> scB, scl
    ln1_scale<<<H_ * S_, 256, 0, stream>>>(scA, scB,
        ln1w + (size_t)l * S_, ln1b + (size_t)l * S_,
        scw + (size_t)l * S_, scb + l, scl);
    // two causal depthwise convs with relu: scB -> scA -> scB
    dwconv_relu<<<dim3(S_ / 32, S_ / 64, H_), 256, 0, stream>>>(scB, scA,
        convw + ((size_t)l * NC_ + 0) * H_ * KW_, convb + ((size_t)l * NC_ + 0) * H_);
    dwconv_relu<<<dim3(S_ / 32, S_ / 64, H_), 256, 0, stream>>>(scA, scB,
        convw + ((size_t)l * NC_ + 1) * H_ * KW_, convb + ((size_t)l * NC_ + 1) * H_);
    // ln2 + mask + softmax + sigmoid(scale) -> f16 probs
    ln2_softmax<<<H_ * S_, 256, 0, stream>>>(scB, p16,
        ln2w + (size_t)l * S_, ln2b + (size_t)l * S_, scl);
    // o = probs @ v  (batched per head, f16 out at head column offset)
    gemm(p16, (long long)S_ * S_, S_, vT16, (long long)HD_ * S_, S_,
         nullptr, a16, (long long)HD_, D_, S_, HD_, S_, H_, 1.f, 0);
    // x += o @ Wo
    gemm(a16, 0, D_, WoT + (size_t)l * D_ * D_, 0, D_, x, nullptr, 0, D_,
         S_, D_, D_, 1, 1.f, 1);
    // MLP
    rmsnorm_f16<<<S_, 256, 0, stream>>>(x, rms2 + (size_t)l * D_, a16);
    gemm(a16, 0, D_, WgT + (size_t)l * FF_ * D_, 0, D_, Gf, nullptr, 0, FF_,
         S_, FF_, D_, 1, 1.f, 0);
    gemm(a16, 0, D_, WuT + (size_t)l * FF_ * D_, 0, D_, Uf, nullptr, 0, FF_,
         S_, FF_, D_, 1, 1.f, 0);
    silu_mul<<<(int)(((size_t)S_ * FF_) / 256), 256, 0, stream>>>(Gf, Uf, a16);
    // x += P @ Wdown
    gemm(a16, 0, FF_, WdT + (size_t)l * D_ * FF_, 0, FF_, x, nullptr, 0, D_,
         S_, D_, FF_, 1, 1.f, 1);
  }
}
